// DistilBertLayerWithMoE_82317343195599
// MI455X (gfx1250) — matrix operand order repaired and checked by reading, not compiled, not run on hardware
//
#include <hip/hip_runtime.h>
#include <hip/hip_bf16.h>
#include <stdint.h>

// ---------------- problem constants ----------------
#define BB   16
#define SS   512
#define DD   768
#define HH   3072
#define NHH  12
#define EE   7
#define DPH  64
#define TOK  (BB*SS)          // 8192
#define OUT0 ((long)TOK*DD)   // 6291456 floats of `out`
// gate weights follow at d_out + OUT0 (TOK*EE floats)

typedef __bf16 bf16;
typedef __attribute__((ext_vector_type(16))) __bf16 v16bf;
typedef __attribute__((ext_vector_type(8)))  float  v8f;

// ---------------- CDNA5 async global->LDS copy (ASYNCcnt path) ----------------
// VDST = per-lane LDS byte offset, VADDR = per-lane 64-bit global address.
__device__ __forceinline__ void async_load_b128(uint32_t lds_off, const void* gptr) {
  asm volatile("global_load_async_to_lds_b128 %0, %1, off"
               :: "v"(lds_off), "v"((uint64_t)(uintptr_t)gptr)
               : "memory");
}

#if defined(__has_builtin)
#  if __has_builtin(__builtin_amdgcn_s_wait_asynccnt)
#    define WAIT_ASYNC0() __builtin_amdgcn_s_wait_asynccnt(0)
#  endif
#endif
#ifndef WAIT_ASYNC0
#  define WAIT_ASYNC0() asm volatile("s_wait_asynccnt 0" ::: "memory")
#endif

// ---------------- WMMA helpers ----------------
__device__ __forceinline__ v8f wmma_bf16(v16bf a, v16bf b, v8f c) {
  // D = A(16x32, bf16) x B(32x16, bf16) + C(16x16, f32)
  return __builtin_amdgcn_wmma_f32_16x16x32_bf16(
      /*neg_a=*/false, a, /*neg_b=*/false, b,
      /*c_mod=*/(short)0, c, /*reuse_a=*/false, /*reuse_b=*/false);
}

// A-matrix fragment (16x32). rowPtr points at (m_row, k_base) with k contiguous.
// Lanes 0-15 hold K 0-15; lanes 16-31 hold K 16-31 (matches B fragments).
__device__ __forceinline__ v16bf load_a_frag(const bf16* rowPtr, int lane) {
  int c0 = (lane < 16) ? 0 : 16;
  v16bf v;
#pragma unroll
  for (int i = 0; i < 8; ++i) {
    v[i]     = rowPtr[c0 + i];
    v[8 + i] = rowPtr[c0 + 8 + i];
  }
  return v;
}

// B-matrix fragment (32x16) with K contiguous in memory (i.e. B^T row-major).
// Same k order as load_a_frag: lanes 0-15 hold K 0-15; lanes 16-31 hold K 16-31; n = lane&15.
__device__ __forceinline__ v16bf load_b_frag_contig(const bf16* colPtr, int lane) {
  int c0 = (lane < 16) ? 0 : 16;
  v16bf v;
#pragma unroll
  for (int i = 0; i < 16; ++i) v[i] = colPtr[c0 + i];
  return v;
}

// B-matrix fragment with K strided (row-major B, stride = row length).
__device__ __forceinline__ v16bf load_b_frag_strided(const bf16* base, long stride, int lane) {
  int c0 = (lane < 16) ? 0 : 16;
  v16bf v;
#pragma unroll
  for (int i = 0; i < 16; ++i) v[i] = base[(long)(c0 + i) * stride];
  return v;
}

__device__ __forceinline__ v8f v8f_zero() {
  v8f z = {0.f,0.f,0.f,0.f,0.f,0.f,0.f,0.f};
  return z;
}

// ---------------- elementwise utility kernels ----------------
__global__ void __launch_bounds__(256)
cvt_f32_bf16_kernel(const float* __restrict__ src, bf16* __restrict__ dst, long n) {
  long i = (long)blockIdx.x * 256 + threadIdx.x;
  long stride = (long)gridDim.x * 256;
  for (; i < n; i += stride) dst[i] = (bf16)src[i];
}

__global__ void __launch_bounds__(256)
copy_f32_kernel(const float* __restrict__ src, float* __restrict__ dst, long n) {
  long i = (long)blockIdx.x * 256 + threadIdx.x;
  long stride = (long)gridDim.x * 256;
  for (; i < n; i += stride) dst[i] = src[i];
}

// ---------------- LayerNorm (one row per block) -> bf16 ----------------
__global__ void __launch_bounds__(256)
ln_bf16_kernel(const float* __restrict__ x, const float* __restrict__ g,
               const float* __restrict__ bta, bf16* __restrict__ y) {
  const int D = DD;
  long row = blockIdx.x;
  const float* xr = x + row * D;
  __shared__ float red[256];
  int tid = threadIdx.x;

  float s = 0.f;
  for (int i = tid; i < D; i += 256) s += xr[i];
  red[tid] = s; __syncthreads();
  for (int st = 128; st > 0; st >>= 1) { if (tid < st) red[tid] += red[tid + st]; __syncthreads(); }
  float mean = red[0] * (1.0f / D);
  __syncthreads();

  float vs = 0.f;
  for (int i = tid; i < D; i += 256) { float d = xr[i] - mean; vs += d * d; }
  red[tid] = vs; __syncthreads();
  for (int st = 128; st > 0; st >>= 1) { if (tid < st) red[tid] += red[tid + st]; __syncthreads(); }
  float inv = rsqrtf(red[0] * (1.0f / D) + 1e-12f);

  for (int i = tid; i < D; i += 256)
    y[row * D + i] = (bf16)(g[i] * (xr[i] - mean) * inv + bta[i]);
}

// ---------------- generic WMMA GEMM: C[M,N] = A[M,K] @ B[K,N], epilogue by MODE
// Double-buffered LDS staging; A tile copied with global_load_async_to_lds_b128
// (ASYNCcnt), B tile transposed manually. One s_wait_asynccnt + one barrier per
// K-step; copy of tile k+1 overlaps the WMMAs on tile k.
// MODE 1: outB = bf16( alpha * (acc + bias[n]) )          (Q/K/V projections)
// MODE 2: outF = acc + bias[n] + res[m*N+n]               (O-proj + residual)
// MODE 3: outB = bf16( gelu(acc + bias[n]) )              (MoE w1)
// MODE 4: outF[m*N+n] += gate[m*gateStride]*(acc+bias[n]) (MoE w2 accumulate)
template<int MODE>
__global__ void __launch_bounds__(256)
gemm_bf16_kernel(const bf16* __restrict__ A, const bf16* __restrict__ B,
                 const float* __restrict__ bias, const float* __restrict__ res,
                 const float* __restrict__ gate, int gateStride, float alpha,
                 float* __restrict__ outF, bf16* __restrict__ outB,
                 int M, int N, int K) {
  __shared__ bf16 sA[2][128][32];   // 16 KB, A tiles (row-major, K contiguous)
  __shared__ bf16 sBt[2][64][32];   // 8 KB,  B tiles transposed (K contiguous)

  int tid  = threadIdx.x;
  int w    = tid >> 5, lane = tid & 31, ln = lane & 15;
  long row0 = (long)blockIdx.y * 128;
  int  col0 = blockIdx.x * 64;
  int wm = w >> 1, wn = w & 1;            // 4x2 wave grid -> 32x32 per wave

  // stage tile at k-offset kb into LDS buffer `buf`
  auto stage = [&](int buf, long kb) {
    // A: thread t -> row t>>1, 16 bf16 (32 B) at (t&1)*16: two async b128s
    int ar = tid >> 1, ac = (tid & 1) * 16;
    const bf16* g = A + (row0 + ar) * (long)K + kb + ac;
    async_load_b128((uint32_t)(uintptr_t)&sA[buf][ar][ac],     g);
    async_load_b128((uint32_t)(uintptr_t)&sA[buf][ar][ac + 8], g + 8);
    // B (transposed into LDS): thread t -> k row t>>3, 8 contiguous cols
    int kr = tid >> 3, c8 = (tid & 7) * 8;
    const bf16* src = B + (kb + kr) * (long)N + col0 + c8;
#pragma unroll
    for (int i = 0; i < 8; ++i) sBt[buf][c8 + i][kr] = src[i];
  };

  v8f acc[2][2];
#pragma unroll
  for (int i = 0; i < 2; ++i)
#pragma unroll
    for (int j = 0; j < 2; ++j) acc[i][j] = v8f_zero();

  stage(0, 0);
  WAIT_ASYNC0();
  __syncthreads();

  int cur = 0;
  for (long kb = 0; kb < K; kb += 32) {
    if (kb + 32 < K) stage(cur ^ 1, kb + 32);   // prefetch next tile (async)

    v16bf a0 = load_a_frag(&sA[cur][wm * 32 + ln][0], lane);
    v16bf a1 = load_a_frag(&sA[cur][wm * 32 + 16 + ln][0], lane);
    v16bf b0 = load_b_frag_contig(&sBt[cur][wn * 32 + ln][0], lane);
    v16bf b1 = load_b_frag_contig(&sBt[cur][wn * 32 + 16 + ln][0], lane);

    acc[0][0] = wmma_bf16(a0, b0, acc[0][0]);
    acc[0][1] = wmma_bf16(a0, b1, acc[0][1]);
    acc[1][0] = wmma_bf16(a1, b0, acc[1][0]);
    acc[1][1] = wmma_bf16(a1, b1, acc[1][1]);

    WAIT_ASYNC0();        // next tile's async copies have landed
    __syncthreads();      // everyone done reading `cur`, writes visible
    cur ^= 1;
  }

  // ---- epilogue; C/D layout: m = r + (lane<16?0:8), n = lane&15 ----
  int mb = (lane < 16) ? 0 : 8;
#pragma unroll
  for (int mi = 0; mi < 2; ++mi)
#pragma unroll
    for (int ni = 0; ni < 2; ++ni) {
      long gmBase = row0 + wm * 32 + mi * 16 + mb;
      int  gn     = col0 + wn * 32 + ni * 16 + ln;
      float bv = bias[gn];
#pragma unroll
      for (int r = 0; r < 8; ++r) {
        long gm = gmBase + r;
        float vv = acc[mi][ni][r] + bv;
        if (MODE == 1) {
          outB[gm * N + gn] = (bf16)(alpha * vv);
        } else if (MODE == 2) {
          outF[gm * N + gn] = vv + res[gm * N + gn];
        } else if (MODE == 3) {
          float ge = 0.5f * vv * (1.0f + erff(vv * 0.70710678118654752f));
          outB[gm * N + gn] = (bf16)ge;
        } else if (MODE == 4) {
          outF[gm * N + gn] += gate[gm * gateStride] * vv;
        }
      }
    }
}

// ---------------- fused attention: one block per (b, h, 32-row q tile) -----
// scores via WMMA -> 64KB fp32 LDS -> masked softmax (shfl reductions) ->
// in-place bf16 probs (register-staged) -> ctx via WMMA -> bf16 ctx buffer.
__global__ void __launch_bounds__(256)
attn_kernel(const bf16* __restrict__ q, const bf16* __restrict__ k,
            const bf16* __restrict__ v, const int* __restrict__ mask,
            bf16* __restrict__ ctx) {
  __shared__ float sSc[32][512];          // 64 KB scores
  bf16* sP = (bf16*)&sSc[0][0];           // aliased [32][512] bf16 probs

  int blk = blockIdx.x;
  int qt  = blk & 15;                     // 512/32 q tiles
  int h   = (blk >> 4) % NHH;
  int b   = blk / (16 * NHH);
  int tid = threadIdx.x;
  int w   = tid >> 5, lane = tid & 31, ln = lane & 15;
  int q0  = qt * 32;
  long base = ((long)b * SS) * DD + h * DPH;   // (b, s=0, h, 0)

  // ---------- phase 1: scores[32 x 512] = q_tile @ k^T ----------
  int cb = w * 64;                        // 64 seq-columns per wave
  v8f acc[2][4];
#pragma unroll
  for (int i = 0; i < 2; ++i)
#pragma unroll
    for (int j = 0; j < 4; ++j) acc[i][j] = v8f_zero();

#pragma unroll
  for (int kk = 0; kk < DPH; kk += 32) {
    v16bf a0 = load_a_frag(q + base + (long)(q0 + ln) * DD + kk, lane);
    v16bf a1 = load_a_frag(q + base + (long)(q0 + 16 + ln) * DD + kk, lane);
#pragma unroll
    for (int nt = 0; nt < 4; ++nt) {
      int ncol = cb + nt * 16 + ln;       // seq index of this lane's column
      // k^T: element (k_dph, n_seq) = k[n_seq][k_dph] -> K contiguous
      v16bf bb = load_b_frag_contig(k + base + (long)ncol * DD + kk, lane);
      acc[0][nt] = wmma_bf16(a0, bb, acc[0][nt]);
      acc[1][nt] = wmma_bf16(a1, bb, acc[1][nt]);
    }
  }
  int mb = (lane < 16) ? 0 : 8;
#pragma unroll
  for (int mi = 0; mi < 2; ++mi)
#pragma unroll
    for (int nt = 0; nt < 4; ++nt)
#pragma unroll
      for (int r = 0; r < 8; ++r)
        sSc[mi * 16 + mb + r][cb + nt * 16 + ln] = acc[mi][nt][r];
  __syncthreads();

  // ---------- phase 2: masked softmax; 8 threads per row, 64 cols each -----
  int row = tid >> 3, sub = tid & 7;
  const int* mrow = mask + b * SS;
  float p[64];
  float mx = -3.4e38f;
#pragma unroll
  for (int i = 0; i < 64; ++i) {
    int c = sub * 64 + i;
    float s = (mrow[c] != 0) ? sSc[row][c] : -3.4e38f;
    p[i] = s;
    mx = fmaxf(mx, s);
  }
#pragma unroll
  for (int d = 1; d < 8; d <<= 1) mx = fmaxf(mx, __shfl_xor(mx, d, 32));
  float sum = 0.f;
#pragma unroll
  for (int i = 0; i < 64; ++i) { float e = __expf(p[i] - mx); p[i] = e; sum += e; }
#pragma unroll
  for (int d = 1; d < 8; d <<= 1) sum += __shfl_xor(sum, d, 32);
  float inv = 1.0f / sum;
#pragma unroll
  for (int i = 0; i < 64; ++i) p[i] *= inv;
  __syncthreads();                         // all fp32 score reads complete
#pragma unroll
  for (int i = 0; i < 64; ++i) sP[row * 512 + sub * 64 + i] = (bf16)p[i];
  __syncthreads();

  // ---------- phase 3: ctx[32 x 64] = probs @ v; one 16x16 tile per wave ---
  int mi = w & 1, ni = w >> 1;             // 2 x 4 tile grid over 32x64
  v8f c2 = v8f_zero();
  for (int kk = 0; kk < SS; kk += 32) {
    v16bf a  = load_a_frag(sP + (long)(mi * 16 + ln) * 512 + kk, lane);
    v16bf bb = load_b_frag_strided(
        v + ((long)b * SS + kk) * DD + h * DPH + ni * 16 + ln, DD, lane);
    c2 = wmma_bf16(a, bb, c2);
  }
#pragma unroll
  for (int r = 0; r < 8; ++r)
    ctx[((long)b * SS + q0 + mi * 16 + mb + r) * DD + h * DPH + ni * 16 + ln] =
        (bf16)c2[r];
}

// ---------------- MoE gate: one wave per token; top-2 masked softmax --------
__global__ void __launch_bounds__(32)
gate_kernel(const bf16* __restrict__ x2, const float* __restrict__ gw,
            const float* __restrict__ gb, float* __restrict__ gout) {
  long t = blockIdx.x;
  int lane = threadIdx.x;
  const bf16* xr = x2 + t * DD;
  float acc[EE];
#pragma unroll
  for (int e = 0; e < EE; ++e) acc[e] = 0.f;
  for (int i = lane; i < DD; i += 32) {
    float xv = (float)xr[i];
#pragma unroll
    for (int e = 0; e < EE; ++e) acc[e] += xv * gw[i * EE + e];
  }
#pragma unroll
  for (int e = 0; e < EE; ++e)
#pragma unroll
    for (int d = 1; d < 32; d <<= 1) acc[e] += __shfl_xor(acc[e], d, 32);
  if (lane == 0) {
    float lg[EE];
#pragma unroll
    for (int e = 0; e < EE; ++e) lg[e] = acc[e] + gb[e];
    int i1 = 0;
#pragma unroll
    for (int e = 1; e < EE; ++e) if (lg[e] > lg[i1]) i1 = e;
    int i2 = (i1 == 0) ? 1 : 0;
#pragma unroll
    for (int e = 0; e < EE; ++e) if (e != i1 && lg[e] > lg[i2]) i2 = e;
    float m  = fmaxf(lg[i1], lg[i2]);
    float e1 = __expf(lg[i1] - m);
    float e2 = __expf(lg[i2] - m);
    float inv = 1.0f / (e1 + e2);
#pragma unroll
    for (int e = 0; e < EE; ++e)
      gout[t * EE + e] = (e == i1) ? e1 * inv : (e == i2) ? e2 * inv : 0.0f;
  }
}

// ---------------- host-side orchestration ----------------
extern "C" void kernel_launch(void* const* d_in, const int* in_sizes, int n_in,
                              void* d_out, int out_size, void* d_ws, size_t ws_size,
                              hipStream_t stream) {
  (void)in_sizes; (void)n_in; (void)out_size; (void)ws_size;
  const float* hidden = (const float*)d_in[0];
  const int*   amask  = (const int*)  d_in[1];
  const float* ln1g   = (const float*)d_in[2];
  const float* ln1b   = (const float*)d_in[3];
  const float* wq = (const float*)d_in[4];  const float* bq = (const float*)d_in[5];
  const float* wk = (const float*)d_in[6];  const float* bk = (const float*)d_in[7];
  const float* wv = (const float*)d_in[8];  const float* bv = (const float*)d_in[9];
  const float* wo = (const float*)d_in[10]; const float* bo = (const float*)d_in[11];
  const float* gatew = (const float*)d_in[12];
  const float* gateb = (const float*)d_in[13];
  const float* w1 = (const float*)d_in[14]; const float* b1 = (const float*)d_in[15];
  const float* w2 = (const float*)d_in[16]; const float* b2 = (const float*)d_in[17];
  const float* ln2g = (const float*)d_in[18];
  const float* ln2b = (const float*)d_in[19];

  float* outPtr = (float*)d_out;            // [TOK, D]
  float* gwOut  = outPtr + OUT0;            // [TOK, E]

  // ---- workspace carving (bytes; every size is 256-aligned) ----
  uint8_t* p = (uint8_t*)d_ws;
  const long TD  = (long)TOK * DD;          // 6,291,456 elems
  const long TH  = (long)TOK * HH;          // 25,165,824 elems
  const long DDsz = (long)DD * DD;
  const long W1sz = (long)EE * DD * HH;
  bf16* x1B  = (bf16*)p; p += TD * 2;
  bf16* qB   = (bf16*)p; p += TD * 2;
  bf16* kB   = (bf16*)p; p += TD * 2;
  bf16* vB   = (bf16*)p; p += TD * 2;
  bf16* ctxB = (bf16*)p; p += TD * 2;
  bf16* x2B  = (bf16*)p; p += TD * 2;
  float* attnF = (float*)p; p += TD * 4;
  bf16* wqB  = (bf16*)p; p += DDsz * 2;
  bf16* wkB  = (bf16*)p; p += DDsz * 2;
  bf16* wvB  = (bf16*)p; p += DDsz * 2;
  bf16* woB  = (bf16*)p; p += DDsz * 2;
  bf16* w1B  = (bf16*)p; p += W1sz * 2;
  bf16* w2B  = (bf16*)p; p += W1sz * 2;
  bf16* midB = (bf16*)p; p += TH * 2;

  // ---- 1) convert weights fp32 -> bf16 ----
  cvt_f32_bf16_kernel<<<1024, 256, 0, stream>>>(wq, wqB, DDsz);
  cvt_f32_bf16_kernel<<<1024, 256, 0, stream>>>(wk, wkB, DDsz);
  cvt_f32_bf16_kernel<<<1024, 256, 0, stream>>>(wv, wvB, DDsz);
  cvt_f32_bf16_kernel<<<1024, 256, 0, stream>>>(wo, woB, DDsz);
  cvt_f32_bf16_kernel<<<8192, 256, 0, stream>>>(w1, w1B, W1sz);
  cvt_f32_bf16_kernel<<<8192, 256, 0, stream>>>(w2, w2B, W1sz);

  // ---- 2) LN1 -> bf16 ----
  ln_bf16_kernel<<<TOK, 256, 0, stream>>>(hidden, ln1g, ln1b, x1B);

  // ---- 3) Q/K/V projections (WMMA) ----
  dim3 gP(DD / 64, TOK / 128);              // (12, 64)
  gemm_bf16_kernel<1><<<gP, 256, 0, stream>>>(x1B, wqB, bq, nullptr, nullptr, 0,
                                              0.125f, nullptr, qB, TOK, DD, DD);
  gemm_bf16_kernel<1><<<gP, 256, 0, stream>>>(x1B, wkB, bk, nullptr, nullptr, 0,
                                              1.0f, nullptr, kB, TOK, DD, DD);
  gemm_bf16_kernel<1><<<gP, 256, 0, stream>>>(x1B, wvB, bv, nullptr, nullptr, 0,
                                              1.0f, nullptr, vB, TOK, DD, DD);

  // ---- 4) fused attention (WMMA scores + softmax + WMMA ctx) ----
  attn_kernel<<<BB * NHH * (SS / 32), 256, 0, stream>>>(qB, kB, vB, amask, ctxB);

  // ---- 5) O-projection + residual -> attnF (fp32) ----
  gemm_bf16_kernel<2><<<gP, 256, 0, stream>>>(ctxB, woB, bo, hidden, nullptr, 0,
                                              1.0f, attnF, nullptr, TOK, DD, DD);

  // ---- 6) LN2 -> bf16 ----
  ln_bf16_kernel<<<TOK, 256, 0, stream>>>(attnF, ln2g, ln2b, x2B);

  // ---- 7) gate: logits, top-2 masked softmax -> gwOut ----
  gate_kernel<<<TOK, 32, 0, stream>>>(x2B, gatew, gateb, gwOut);

  // ---- 8) out := attn (residual base for MoE accumulation) ----
  copy_f32_kernel<<<4096, 256, 0, stream>>>(attnF, outPtr, TD);

  // ---- 9) MoE experts: mid = gelu(x2@w1+b1); out += g_e*(mid@w2+b2) ----
  dim3 g1(HH / 64, TOK / 128);              // (48, 64)
  dim3 g2(DD / 64, TOK / 128);              // (12, 64)
  for (int e = 0; e < EE; ++e) {
    gemm_bf16_kernel<3><<<g1, 256, 0, stream>>>(
        x2B, w1B + (long)e * DD * HH, b1 + (long)e * HH, nullptr, nullptr, 0,
        1.0f, nullptr, midB, TOK, HH, DD);
    gemm_bf16_kernel<4><<<g2, 256, 0, stream>>>(
        midB, w2B + (long)e * HH * DD, b2 + (long)e * DD, nullptr,
        gwOut + e, EE, 1.0f, outPtr, nullptr, TOK, DD, HH);
  }
}
